// LocalWindowedAttention_19241453486181
// MI455X (gfx1250) — matrix operand order repaired and checked by reading, not compiled, not run on hardware
//
#include <hip/hip_runtime.h>
#include <hip/hip_bf16.h>

// ---------------------------------------------------------------------------
// LocalWindowedAttention on gfx1250 (MI455X): bf16 WMMA everywhere.
//   B=2, L=2048, D=2048, H=16, HKV=4, DH=128, WINDOW=256, GROUP=4
// ---------------------------------------------------------------------------

typedef __bf16 bf16_t;
typedef __attribute__((ext_vector_type(16))) __bf16 v16bf;
typedef __attribute__((ext_vector_type(8)))  __bf16 v8bf;
typedef __attribute__((ext_vector_type(8)))  float  v8f;

#define BATCH   2
#define SEQ     2048
#define DMODEL  2048
#define NH      16
#define NHKV    4
#define DH      128
#define WIN     256
#define ROWS    (BATCH * SEQ)          // 4096

// ---- workspace layout (bytes) ----
#define OFF_XB    ((size_t)0)           // bf16 [4096,2048]   16 MiB
#define OFF_WQB   ((size_t)16777216)    // bf16 [2048,2048]    8 MiB
#define OFF_WKVB  ((size_t)25165824)    // bf16 [1024,2048]    4 MiB
#define OFF_WOB   ((size_t)29360128)    // bf16 [2048,2048]    8 MiB
#define OFF_QF    ((size_t)37748736)    // f32  [4096,2048]   32 MiB
#define OFF_KVF   ((size_t)71303168)    // f32  [4096,1024]   16 MiB
#define OFF_QB    ((size_t)88080384)    // bf16 [4096,2048]   16 MiB
#define OFF_KB    ((size_t)104857600)   // bf16 [4096,512]     4 MiB
#define OFF_VB    ((size_t)109051904)   // bf16 [4096,512]     4 MiB
#define OFF_OB    ((size_t)113246208)   // bf16 [4096,2048]   16 MiB

__device__ __forceinline__ unsigned short f2bfu(float f) {
    unsigned u = __float_as_uint(f);
    return (unsigned short)((u + 0x7fffu + ((u >> 16) & 1u)) >> 16);
}
__device__ __forceinline__ bf16_t f2bf(float f) {
    unsigned short h = f2bfu(f);
    bf16_t b; __builtin_memcpy(&b, &h, 2); return b;
}

__device__ __forceinline__ v8f wmma_bf16(v16bf a, v16bf b, v8f c) {
    return __builtin_amdgcn_wmma_f32_16x16x32_bf16(
        /*neg_a=*/false, a, /*neg_b=*/false, b,
        /*c_mod=*/(short)0, c, /*reuse_a=*/false, /*reuse_b=*/false);
}

// ---------------------------------------------------------------------------
// Kernel 1: f32 -> bf16 cast
// ---------------------------------------------------------------------------
__global__ void cast_f32_bf16(const float* __restrict__ in,
                              bf16_t* __restrict__ out, int n) {
    int i = (blockIdx.x * blockDim.x + threadIdx.x) * 4;
    if (i + 3 < n) {
        float4 v = *(const float4*)(in + i);
        out[i + 0] = f2bf(v.x);
        out[i + 1] = f2bf(v.y);
        out[i + 2] = f2bf(v.z);
        out[i + 3] = f2bf(v.w);
    } else {
        for (; i < n; ++i) out[i] = f2bf(in[i]);
    }
}

// ---------------------------------------------------------------------------
// Kernel 2: C[M,N] = A[M,K] (bf16) * W[N,K]^T (bf16), f32 out.
// One wave computes a 32x64 tile (2 M-subtiles x 4 N-subtiles = 8 WMMAs per
// k-step of 32).  B fragments are reused across both M-subtiles.
// A-frag (16x32 bf16): lane<16 -> M=lane, K 0-15 contiguous;
//                                  lane>=16 -> K 16-31 contiguous.
// B-frag (32x16 bf16): lane holds column n; lanes 0-15: K 0-15 contiguous,
//                      lanes 16-31: K 16-31 contiguous == W row-major reads.
// ---------------------------------------------------------------------------
__global__ __launch_bounds__(256)
void gemm_bf16_f32(const bf16_t* __restrict__ A, const bf16_t* __restrict__ W,
                   float* __restrict__ C, int M, int N, int K) {
    const int lane = threadIdx.x & 31;
    const int wave = (int)((blockIdx.x * blockDim.x + threadIdx.x) >> 5);
    const int tilesN = N >> 6;
    const int tm = wave / tilesN;
    const int tn = wave % tilesN;
    const int m0 = tm * 32, n0 = tn * 64;
    if (m0 >= M) return;
    const int half = lane >> 4, lm = lane & 15;

    v8f acc[2][4] = {};
    const bf16_t* arow0 = A + (size_t)(m0 + lm) * K;
    const bf16_t* arow1 = A + (size_t)(m0 + 16 + lm) * K;

    for (int k0 = 0; k0 < K; k0 += 32) {
        __builtin_prefetch(arow0 + k0 + 256, 0, 3);
        union { v16bf v; v8bf h[2]; } a0, a1;
        a0.h[0] = *(const v8bf*)(arow0 + k0 + half * 16);
        a0.h[1] = *(const v8bf*)(arow0 + k0 + half * 16 + 8);
        a1.h[0] = *(const v8bf*)(arow1 + k0 + half * 16);
        a1.h[1] = *(const v8bf*)(arow1 + k0 + half * 16 + 8);
#pragma unroll
        for (int j = 0; j < 4; ++j) {
            const bf16_t* wrow =
                W + (size_t)(n0 + j * 16 + lm) * K + k0 + half * 16;
            v16bf b = *(const v16bf*)wrow;
            acc[0][j] = wmma_bf16(a0.v, b, acc[0][j]);
            acc[1][j] = wmma_bf16(a1.v, b, acc[1][j]);
        }
    }
#pragma unroll
    for (int i = 0; i < 2; ++i)
#pragma unroll
        for (int j = 0; j < 4; ++j)
#pragma unroll
            for (int r = 0; r < 8; ++r)
                C[(size_t)(m0 + i * 16 + r + half * 8) * N +
                  (n0 + j * 16 + lm)] = acc[i][j][r];
}

// ---------------------------------------------------------------------------
// Kernel 3: RoPE + L2 qk-norm (+temp) -> bf16.  One wave per (row, task).
// Tasks per row: 16 q-heads, 4 k-heads (rope+norm), 4 v-heads (cast only).
// ---------------------------------------------------------------------------
__global__ __launch_bounds__(256)
void rope_norm(const float* __restrict__ qf, const float* __restrict__ kvf,
               const float* __restrict__ qk_temp,
               bf16_t* __restrict__ qb, bf16_t* __restrict__ kb,
               bf16_t* __restrict__ vb) {
    const int lane = threadIdx.x & 31;
    const int wave = (int)((blockIdx.x * blockDim.x + threadIdx.x) >> 5);
    const int row = wave / 24;
    const int t = wave % 24;
    if (row >= ROWS) return;
    const int pos = row & (SEQ - 1);

    if (t >= 20) {                       // v: plain cast
        const float* src = kvf + (size_t)row * 1024 + 512 + (t - 20) * DH;
        bf16_t* dst = vb + (size_t)row * 512 + (t - 20) * DH;
        int d = lane * 4;
#pragma unroll
        for (int i = 0; i < 4; ++i) dst[d + i] = f2bf(src[d + i]);
        return;
    }

    const float* src;
    bf16_t* dst;
    float temp;
    if (t < 16) {                        // q head t
        src = qf + (size_t)row * DMODEL + t * DH;
        dst = qb + (size_t)row * DMODEL + t * DH;
        temp = qk_temp[t];
    } else {                             // k head t-16
        src = kvf + (size_t)row * 1024 + (t - 16) * DH;
        dst = kb + (size_t)row * 512 + (t - 16) * DH;
        temp = 1.0f;
    }

    const int d0 = lane * 2, d1 = lane * 2 + 1;      // dims in [0,64)
    const float KF = -0.14391157f;                    // -ln(10000)/64
    float ang0 = (float)pos * __expf((float)d0 * KF);
    float ang1 = (float)pos * __expf((float)d1 * KF);
    float c0 = __cosf(ang0), s0 = __sinf(ang0);
    float c1 = __cosf(ang1), s1 = __sinf(ang1);

    float x0 = src[d0], x0b = src[d0 + 64];
    float x1 = src[d1], x1b = src[d1 + 64];
    float y0  = x0 * c0 - x0b * s0;
    float y0b = x0b * c0 + x0 * s0;
    float y1  = x1 * c1 - x1b * s1;
    float y1b = x1b * c1 + x1 * s1;

    float ss = y0 * y0 + y0b * y0b + y1 * y1 + y1b * y1b;
#pragma unroll
    for (int m = 1; m < 32; m <<= 1) ss += __shfl_xor(ss, m, 32);
    float sc = temp / fmaxf(sqrtf(ss), 1e-12f);

    dst[d0] = f2bf(y0 * sc);
    dst[d0 + 64] = f2bf(y0b * sc);
    dst[d1] = f2bf(y1 * sc);
    dst[d1 + 64] = f2bf(y1b * sc);
}

// ---------------------------------------------------------------------------
// Kernel 4: sliding-window flash attention. One wave per 16-query tile.
// Per 32-key chunk:
//   1. 16x GLOBAL_LOAD_ASYNC_TO_LDS_B128 kick off the 8KB V tile copy
//      (ASYNCcnt) -- overlaps with the score computation below.
//   2. S = Q*K^T (8 WMMA), online softmax, P -> LDS (bf16).
//   3. s_wait_asynccnt 0, gather transposed V B-frags from LDS, O += P*V.
// ---------------------------------------------------------------------------
__global__ __launch_bounds__(128)
void attn_window(const bf16_t* __restrict__ qb, const bf16_t* __restrict__ kb,
                 const bf16_t* __restrict__ vb, bf16_t* __restrict__ ob) {
    __shared__ unsigned short ldsP[4][16 * 32];    // P tile, [m][key]
    __shared__ unsigned short ldsV[4][32 * DH];    // V tile, [key][dim]

    const int lane = threadIdx.x & 31;
    const int w = threadIdx.x >> 5;
    const int tile = blockIdx.x * 4 + w;             // 4096 tiles
    if (tile >= BATCH * NH * (SEQ / 16)) return;
    const int b = tile >> 11;
    const int rem = tile & 2047;
    const int h = rem >> 7;
    const int q0 = (rem & 127) << 4;
    const int hkv = h >> 2;
    const int half = lane >> 4, lm = lane & 15;

    unsigned short* P  = ldsP[w];
    unsigned short* VL = ldsV[w];
    const unsigned vbase = (unsigned)(size_t)(void*)VL;  // LDS byte offset

    // Q fragments: 16x128 -> 4 A-frags of 16x32
    const bf16_t* qrow = qb + ((size_t)(b * SEQ + q0 + lm)) * DMODEL + h * DH;
    union { v16bf v; v8bf hh[2]; } qa[4];
#pragma unroll
    for (int kk = 0; kk < 4; ++kk) {
        qa[kk].hh[0] = *(const v8bf*)(qrow + kk * 32 + half * 16);
        qa[kk].hh[1] = *(const v8bf*)(qrow + kk * 32 + half * 16 + 8);
    }

    v8f o[8] = {};
    float mstate[8], lstate[8];
#pragma unroll
    for (int r = 0; r < 8; ++r) { mstate[r] = -3.0e38f; lstate[r] = 0.0f; }

    int cstart = q0 - (WIN - 1);
    if (cstart < 0) cstart = 0;
    cstart &= ~31;

    for (int c0 = cstart; c0 <= q0 + 15; c0 += 32) {
        // ---- async copy of V tile (32 keys x 128 dims bf16) into LDS ----
        // Each instruction: 32 lanes x 16B = 2 key rows (256B each).
#pragma unroll
        for (int it = 0; it < 16; ++it) {
            const int keyl = it * 2 + half;
            unsigned laddr = vbase + (unsigned)(keyl * 256 + lm * 16);
            const bf16_t* gp = vb +
                ((size_t)(b * SEQ + c0 + keyl)) * 512 + hkv * DH + lm * 8;
            asm volatile("global_load_async_to_lds_b128 %0, %1, off"
                         :: "v"(laddr), "v"(gp) : "memory");
        }

        // ---- S = Q * K^T (16 x 32 scores) ----
        v8f s[2] = {};
#pragma unroll
        for (int kk = 0; kk < 4; ++kk) {
#pragma unroll
            for (int j = 0; j < 2; ++j) {
                const bf16_t* krow = kb +
                    ((size_t)(b * SEQ + c0 + j * 16 + lm)) * 512 +
                    hkv * DH + kk * 32 + half * 16;
                v16bf bf = *(const v16bf*)krow;
                s[j] = wmma_bf16(qa[kk].v, bf, s[j]);
            }
        }

        // ---- mask + online softmax (row = r + 8*half, col lane%16) ----
        float sc[8];
#pragma unroll
        for (int r = 0; r < 8; ++r) {
            int m = q0 + r + half * 8;               // absolute query
            int k0i = c0 + lm, k1i = c0 + 16 + lm;   // absolute keys
            bool a0 = (k0i <= m) && (m - k0i < WIN);
            bool a1 = (k1i <= m) && (m - k1i < WIN);
            float v0 = s[0][r], v1 = s[1][r];
            float rmax = fmaxf(a0 ? v0 : -3.0e38f, a1 ? v1 : -3.0e38f);
#pragma unroll
            for (int msk = 1; msk < 16; msk <<= 1)
                rmax = fmaxf(rmax, __shfl_xor(rmax, msk, 16));
            float newm = fmaxf(mstate[r], rmax);
            float scale = __expf(mstate[r] - newm);
            mstate[r] = newm;
            float p0 = a0 ? __expf(v0 - newm) : 0.0f;
            float p1 = a1 ? __expf(v1 - newm) : 0.0f;
            float rs = p0 + p1;
#pragma unroll
            for (int msk = 1; msk < 16; msk <<= 1)
                rs += __shfl_xor(rs, msk, 16);
            lstate[r] = lstate[r] * scale + rs;
            sc[r] = scale;
            int mm = r + half * 8;
            P[mm * 32 + lm] = f2bfu(p0);
            P[mm * 32 + 16 + lm] = f2bfu(p1);
        }
#pragma unroll
        for (int f = 0; f < 8; ++f)
#pragma unroll
            for (int r = 0; r < 8; ++r) o[f][r] *= sc[r];

        // ---- P as A-frag from LDS (DS ops are in-order per wave) ----
        asm volatile("s_wait_dscnt 0" ::: "memory");
        union { v16bf v; unsigned u[8]; } pa;
#pragma unroll
        for (int vv = 0; vv < 8; ++vv) {
            int kk = 2 * vv + half * 16;
            pa.u[vv] = *(const unsigned*)&P[lm * 32 + kk];
        }

        // ---- V B-frags: transpose during LDS gather; O += P * V ----
        asm volatile("s_wait_asynccnt 0" ::: "memory");
#pragma unroll
        for (int f = 0; f < 8; ++f) {
            union { v16bf v; unsigned u[8]; } vbf;
            int d = f * 16 + lm;
#pragma unroll
            for (int vv = 0; vv < 8; ++vv) {
                int kk = 2 * vv + half * 16;
                unsigned lo = VL[kk * DH + d];
                unsigned hi = VL[(kk + 1) * DH + d];
                vbf.u[vv] = lo | (hi << 16);
            }
            o[f] = wmma_bf16(pa.v, vbf.v, o[f]);
        }
    }

    // ---- finalize: divide by l, write bf16 ----
#pragma unroll
    for (int r = 0; r < 8; ++r) {
        float inv = 1.0f / lstate[r];
        size_t grow = (size_t)(b * SEQ + q0 + r + half * 8);
#pragma unroll
        for (int f = 0; f < 8; ++f)
            ob[grow * DMODEL + h * DH + f * 16 + lm] = f2bf(o[f][r] * inv);
    }
}

// ---------------------------------------------------------------------------
extern "C" void kernel_launch(void* const* d_in, const int* in_sizes, int n_in,
                              void* d_out, int out_size, void* d_ws,
                              size_t ws_size, hipStream_t stream) {
    const float* x    = (const float*)d_in[0];
    const float* Wq   = (const float*)d_in[1];
    const float* Wkv  = (const float*)d_in[2];
    const float* Wo   = (const float*)d_in[3];
    const float* temp = (const float*)d_in[4];

    char* ws = (char*)d_ws;
    bf16_t* xb   = (bf16_t*)(ws + OFF_XB);
    bf16_t* Wqb  = (bf16_t*)(ws + OFF_WQB);
    bf16_t* Wkvb = (bf16_t*)(ws + OFF_WKVB);
    bf16_t* Wob  = (bf16_t*)(ws + OFF_WOB);
    float*  qf   = (float*) (ws + OFF_QF);
    float*  kvf  = (float*) (ws + OFF_KVF);
    bf16_t* qb   = (bf16_t*)(ws + OFF_QB);
    bf16_t* kb   = (bf16_t*)(ws + OFF_KB);
    bf16_t* vb   = (bf16_t*)(ws + OFF_VB);
    bf16_t* ob   = (bf16_t*)(ws + OFF_OB);

    auto cast_n = [&](const float* src, bf16_t* dst, int n) {
        int blocks = (n / 4 + 255) / 256;
        cast_f32_bf16<<<blocks, 256, 0, stream>>>(src, dst, n);
    };
    cast_n(x,   xb,   ROWS * DMODEL);        // 8,388,608
    cast_n(Wq,  Wqb,  DMODEL * DMODEL);      // 4,194,304
    cast_n(Wkv, Wkvb, 1024 * DMODEL);        // 2,097,152
    cast_n(Wo,  Wob,  DMODEL * DMODEL);

    // q = x @ Wq^T : [4096,2048] ; kv = x @ Wkv^T : [4096,1024]
    gemm_bf16_f32<<<(ROWS / 32) * (DMODEL / 64) / 8, 256, 0, stream>>>(
        xb, Wqb, qf, ROWS, DMODEL, DMODEL);
    gemm_bf16_f32<<<(ROWS / 32) * (1024 / 64) / 8, 256, 0, stream>>>(
        xb, Wkvb, kvf, ROWS, 1024, DMODEL);

    // RoPE + qk-norm + casts: 4096 rows * 24 wave-tasks / 8 waves per block
    rope_norm<<<ROWS * 24 / 8, 256, 0, stream>>>(qf, kvf, temp, qb, kb, vb);

    // windowed attention: 4096 query tiles, 4 waves per block
    attn_window<<<BATCH * NH * (SEQ / 16) / 4, 128, 0, stream>>>(qb, kb, vb, ob);

    // out = attn @ Wo^T -> d_out (f32 [4096,2048])
    gemm_bf16_f32<<<(ROWS / 32) * (DMODEL / 64) / 8, 256, 0, stream>>>(
        ob, Wob, (float*)d_out, ROWS, DMODEL, DMODEL);
}
